// ManualLSTM_13194139533664
// MI455X (gfx1250) — matrix operand order repaired and checked by reading, not compiled, not run on hardware
//
#include <hip/hip_runtime.h>
#include <hip/hip_bf16.h>

// Problem constants (match reference)
#define BB 32     // batch
#define SS 512    // seq len
#define II 512    // input size
#define HH 512    // hidden
#define GG 2048   // 4*H gate width
#define NWG 32    // workgroups in persistent recurrent kernel
#define TPB 256   // threads per block (8 wave32)
#define HPAD 520  // LDS row stride for h staging (bank-conflict padding, 16B aligned)

typedef unsigned int u32;
typedef unsigned short u16;
typedef __attribute__((ext_vector_type(16))) __bf16 v16bf;
typedef __attribute__((ext_vector_type(8)))  float  v8f;

union Frag16 { uint4 q[2]; v16bf v; };

__device__ inline u16 f2bf(float f) {
  u32 x = __float_as_uint(f);
  u32 r = (x + 0x7FFFu + ((x >> 16) & 1u)) >> 16;  // round-to-nearest-even
  return (u16)r;
}

__device__ inline float sigmoidf_(float x) { return 1.0f / (1.0f + __expf(-x)); }

// ---------------- utility kernels ----------------

__global__ void zero_kernel(u32* __restrict__ p, int n) {
  for (int t = blockIdx.x * blockDim.x + threadIdx.x; t < n; t += gridDim.x * blockDim.x)
    p[t] = 0u;
}

// x fp32 [B][S][I] -> bf16 [S][B][I]
__global__ void convert_x_kernel(const float* __restrict__ x, u16* __restrict__ xbf) {
  const int total = BB * SS * II;
  for (int t = blockIdx.x * blockDim.x + threadIdx.x; t < total; t += gridDim.x * blockDim.x) {
    int i = t % II; int rest = t / II; int s = rest % SS; int b = rest / SS;
    xbf[(s * BB + b) * II + i] = f2bf(x[t]);
  }
}

// w fp32 [N][K] row-major (i.e. [4H, in]); pack B-matrix fragments for B[k][n] = w[n][k].
// Fragment layout (16x16x32 bf16 WMMA, B is K=32 x N=16):
//   lane<16 : N=lane,    K = kt*32 + 2j,2j+1   (j=0..7)
//   lane>=16: N=lane-16, K = kt*32 + 16+2j,...
// pack dword index: ((kt*(N/16)+nt)*32 + lane)*8 + j
__global__ void pack_w_kernel(const float* __restrict__ w, u32* __restrict__ bp, int K, int N) {
  const int total = (K >> 1) * N;
  const int ntiles = N >> 4;
  for (int t = blockIdx.x * blockDim.x + threadIdx.x; t < total; t += gridDim.x * blockDim.x) {
    int j = t & 7;
    int lane = (t >> 3) & 31;
    int rest = t >> 8;
    int nt = rest % ntiles;
    int kt = rest / ntiles;
    int n = (nt << 4) + (lane & 15);
    int k = (kt << 5) + ((lane >> 4) << 4) + (j << 1);
    u32 lo = (u32)f2bf(w[n * K + k]);
    u32 hi = (u32)f2bf(w[n * K + k + 1]);
    bp[t] = lo | (hi << 16);
  }
}

__global__ void bias_combine_kernel(const float* __restrict__ a, const float* __restrict__ b,
                                    float* __restrict__ o, int n) {
  int t = blockIdx.x * blockDim.x + threadIdx.x;
  if (t < n) o[t] = a[t] + b[t];
}

// ---------------- WMMA GEMM: C[M,N] = A[M,K](bf16 row-major) * Bpack + bias[n] ----------------
// block = 8 waves arranged 4(M) x 2(N); wave tile 16M x 64N; block tile 64M x 128N.
__global__ void __launch_bounds__(TPB) wmma_gemm_bias_kernel(
    const u16* __restrict__ A, const u32* __restrict__ Bp,
    const float* __restrict__ bias, float* __restrict__ C,
    int M, int N, int K) {
  (void)M;
  const int lane = threadIdx.x & 31;
  const int wave = threadIdx.x >> 5;
  const int wm = wave >> 1, wn = wave & 1;
  const int m0 = blockIdx.y * 64 + wm * 16;
  const int n0 = blockIdx.x * 128 + wn * 64;
  const int ntiles = N >> 4;
  const int mr = lane & 15;
  const int hf = lane >> 4;

  v8f acc[4] = {};
  const uint4* Bq = (const uint4*)Bp;
  const int kts = K >> 5;
  for (int kt = 0; kt < kts; ++kt) {
    Frag16 a;
    const u16* arow = A + (size_t)(m0 + mr) * K + (kt << 5) + hf * 16;
    a.q[0] = *(const uint4*)(arow);
    a.q[1] = *(const uint4*)(arow + 8);
#pragma unroll
    for (int t = 0; t < 4; ++t) {
      const int ntg = (n0 >> 4) + t;
      const uint4* bq = Bq + ((((size_t)kt * ntiles + ntg) * 32 + lane) << 1);
      Frag16 b;
      b.q[0] = bq[0];
      b.q[1] = bq[1];
      acc[t] = __builtin_amdgcn_wmma_f32_16x16x32_bf16(false, a.v, false, b.v,
                                                       (short)0, acc[t], false, false);
    }
  }
  const int rbase = m0 + (hf ? 8 : 0);
#pragma unroll
  for (int t = 0; t < 4; ++t) {
    const int col = n0 + t * 16 + mr;
    const float bv = bias[col];
#pragma unroll
    for (int j = 0; j < 8; ++j)
      C[(size_t)(rbase + j) * N + col] = acc[t][j] + bv;
  }
}

// ---------------- persistent bidirectional LSTM layer ----------------

__device__ inline void grid_barrier(u32* cnt, u32 target) {
  __threadfence();
  __syncthreads();
  if (threadIdx.x == 0) {
    __hip_atomic_fetch_add(cnt, 1u, __ATOMIC_RELAXED, __HIP_MEMORY_SCOPE_AGENT);
    while (__hip_atomic_load(cnt, __ATOMIC_RELAXED, __HIP_MEMORY_SCOPE_AGENT) < target) {
      __builtin_amdgcn_s_sleep(2);
    }
  }
  __syncthreads();
  __threadfence();
}

// grid = NWG blocks * TPB threads = 256 waves. Wave gw: dir = gw>>7 (uniform per block),
// gate tile nt = gw&127. W_hh fragments for the wave's tile are preloaded into 128 VGPRs
// and stay register-resident for all 512 steps. h is staged block-wide in LDS each step.
__global__ void __launch_bounds__(TPB) lstm_layer_kernel(
    const float* __restrict__ xp_f, const float* __restrict__ xp_b,
    const u32* __restrict__ whh_f, const u32* __restrict__ whh_b,
    float* __restrict__ gates,      // [2][B][G] fp32 scratch
    u16* __restrict__ hbf,          // [2][B*H] bf16 recurrent state
    float* __restrict__ cst,        // [2][B*H] fp32 cell state
    u16* __restrict__ seq_out,      // [S][B][2H] bf16 (layer 0) or nullptr
    float* __restrict__ out_f32,    // [B][S][2H] fp32 (layer 1) or nullptr
    float* __restrict__ dout_h, float* __restrict__ dout_c,  // [2][B][H] final states
    u32* __restrict__ cnt) {
  __shared__ __align__(16) u16 hsh[BB * HPAD];  // 33,280 B: block's direction h matrix

  const int lane = threadIdx.x & 31;
  const int wave = threadIdx.x >> 5;
  const int gw = blockIdx.x * 8 + wave;   // 0..255
  const int dir = gw >> 7;                // uniform per block
  const int nt = gw & 127;
  const int n0 = nt << 4;
  const int mr = lane & 15;
  const int hf = lane >> 4;
  const int rb = hf ? 8 : 0;

  const float* xp = dir ? xp_b : xp_f;
  const uint4* wq = (const uint4*)(dir ? whh_b : whh_f);
  const u16* hsrc = hbf + dir * (BB * HH);
  float* gdst = gates + dir * (BB * GG);
  const int tidg = blockIdx.x * TPB + threadIdx.x;  // 0..8191
  u32 barno = 0;

  // Preload this wave's W_hh^T gate-tile fragments: 16 K-tiles x 32B = 128 VGPRs/lane.
  Frag16 wreg[16];
#pragma unroll
  for (int kt = 0; kt < (HH >> 5); ++kt) {
    const uint4* bq = wq + (((kt * 128 + nt) * 32 + lane) << 1);
    wreg[kt].q[0] = bq[0];
    wreg[kt].q[1] = bq[1];
  }

  // h staging copy assignments: 8 threads per row, 64 halves (8 uint4) per thread.
  const int crow = threadIdx.x >> 3;
  const int ccol = (threadIdx.x & 7) * 64;

  for (int step = 0; step < SS; ++step) {
    const int s_d = dir ? (SS - 1 - step) : step;

    // ---- stage h (bf16 [32][512]) into LDS ----
    {
      const uint4* gsrc = (const uint4*)(hsrc + crow * HH + ccol);
      uint4* ldst = (uint4*)(&hsh[crow * HPAD + ccol]);
#pragma unroll
      for (int i = 0; i < 8; ++i) ldst[i] = gsrc[i];
    }
    __syncthreads();

    // ---- GEMM phase: two 16-row tiles (b=0..15, b=16..31), 16 gate cols, K=512 ----
    v8f acc0 = {}, acc1 = {};
    {
      const float* xpr = xp + (size_t)s_d * (BB * GG) + n0 + mr;
#pragma unroll
      for (int j = 0; j < 8; ++j) {
        acc0[j] = xpr[(rb + j) * GG];
        acc1[j] = xpr[(16 + rb + j) * GG];
      }
    }
#pragma unroll
    for (int kt = 0; kt < (HH >> 5); ++kt) {
      Frag16 a0, a1;
      const u16* ar0 = &hsh[mr * HPAD + (kt << 5) + hf * 16];
      a0.q[0] = *(const uint4*)(ar0);
      a0.q[1] = *(const uint4*)(ar0 + 8);
      const u16* ar1 = ar0 + 16 * HPAD;
      a1.q[0] = *(const uint4*)(ar1);
      a1.q[1] = *(const uint4*)(ar1 + 8);
      acc0 = __builtin_amdgcn_wmma_f32_16x16x32_bf16(false, a0.v, false, wreg[kt].v,
                                                     (short)0, acc0, false, false);
      acc1 = __builtin_amdgcn_wmma_f32_16x16x32_bf16(false, a1.v, false, wreg[kt].v,
                                                     (short)0, acc1, false, false);
    }
    {
      float* gp = gdst + n0 + mr;
#pragma unroll
      for (int j = 0; j < 8; ++j) {
        gp[(rb + j) * GG] = acc0[j];
        gp[(16 + rb + j) * GG] = acc1[j];
      }
    }
    ++barno; grid_barrier(cnt, barno * NWG);

    // ---- cell phase: 2 dirs * 32 b * 512 j = 32768 tasks over 8192 threads ----
#pragma unroll
    for (int t = 0; t < 4; ++t) {
      const int idx = tidg + t * (NWG * TPB);
      const int d2 = idx >> 14;
      const int rem = idx & 16383;
      const int b = rem >> 9;
      const int j = rem & 511;
      const int sd2 = d2 ? (SS - 1 - step) : step;
      const float* gg = gates + d2 * (BB * GG) + b * GG + j;
      const float iv = sigmoidf_(gg[0]);
      const float fv = sigmoidf_(gg[HH]);
      const float gv = tanhf(gg[2 * HH]);
      const float ov = sigmoidf_(gg[3 * HH]);
      const int ci = d2 * (BB * HH) + b * HH + j;
      const float c = fv * cst[ci] + iv * gv;
      cst[ci] = c;
      const float h = ov * tanhf(c);
      hbf[ci] = f2bf(h);
      if (seq_out) seq_out[(size_t)(sd2 * BB + b) * (2 * HH) + d2 * HH + j] = f2bf(h);
      if (out_f32) out_f32[(size_t)(b * SS + sd2) * (2 * HH) + d2 * HH + j] = h;
      if (step == SS - 1) { dout_h[ci] = h; dout_c[ci] = c; }
    }
    ++barno; grid_barrier(cnt, barno * NWG);
  }
}

// ---------------- host orchestration ----------------

extern "C" void kernel_launch(void* const* d_in, const int* in_sizes, int n_in,
                              void* d_out, int out_size, void* d_ws, size_t ws_size,
                              hipStream_t stream) {
  (void)in_sizes; (void)n_in; (void)out_size; (void)ws_size;
  const float* x = (const float*)d_in[0];
  const float* w_ih[4] = { (const float*)d_in[1], (const float*)d_in[5],
                           (const float*)d_in[9], (const float*)d_in[13] };
  const float* w_hh[4] = { (const float*)d_in[2], (const float*)d_in[6],
                           (const float*)d_in[10], (const float*)d_in[14] };
  const float* b_ih[4] = { (const float*)d_in[3], (const float*)d_in[7],
                           (const float*)d_in[11], (const float*)d_in[15] };
  const float* b_hh[4] = { (const float*)d_in[4], (const float*)d_in[8],
                           (const float*)d_in[12], (const float*)d_in[16] };
  float* out = (float*)d_out;

  // workspace carve-out (256-byte aligned)
  char* ws = (char*)d_ws;
  size_t off = 0;
  auto alloc = [&](size_t bytes) -> char* {
    char* p = ws + off;
    off = (off + bytes + 255) & ~(size_t)255;
    return p;
  };
  u16* xbf   = (u16*)alloc((size_t)SS * BB * II * 2);
  u16* seq1  = (u16*)alloc((size_t)SS * BB * 2 * HH * 2);
  u32* wih_p[4];
  wih_p[0] = (u32*)alloc((size_t)II * GG * 2);
  wih_p[1] = (u32*)alloc((size_t)II * GG * 2);
  wih_p[2] = (u32*)alloc((size_t)2 * HH * GG * 2);
  wih_p[3] = (u32*)alloc((size_t)2 * HH * GG * 2);
  u32* whh_p[4];
  for (int i = 0; i < 4; ++i) whh_p[i] = (u32*)alloc((size_t)HH * GG * 2);
  float* biasc[4];
  for (int i = 0; i < 4; ++i) biasc[i] = (float*)alloc((size_t)GG * 4);
  float* xpf   = (float*)alloc((size_t)SS * BB * GG * 4);
  float* xpb   = (float*)alloc((size_t)SS * BB * GG * 4);
  float* gates = (float*)alloc((size_t)2 * BB * GG * 4);
  u16*   hbf   = (u16*)alloc((size_t)2 * BB * HH * 2);
  float* cst   = (float*)alloc((size_t)2 * BB * HH * 4);
  u32*   cnt   = (u32*)alloc(256);

  // conversions + weight packing
  convert_x_kernel<<<2048, TPB, 0, stream>>>(x, xbf);
  pack_w_kernel<<<2048, TPB, 0, stream>>>(w_ih[0], wih_p[0], II, GG);
  pack_w_kernel<<<2048, TPB, 0, stream>>>(w_ih[1], wih_p[1], II, GG);
  pack_w_kernel<<<4096, TPB, 0, stream>>>(w_ih[2], wih_p[2], 2 * HH, GG);
  pack_w_kernel<<<4096, TPB, 0, stream>>>(w_ih[3], wih_p[3], 2 * HH, GG);
  for (int i = 0; i < 4; ++i)
    pack_w_kernel<<<2048, TPB, 0, stream>>>(w_hh[i], whh_p[i], HH, GG);
  for (int i = 0; i < 4; ++i)
    bias_combine_kernel<<<GG / TPB, TPB, 0, stream>>>(b_ih[i], b_hh[i], biasc[i], GG);

  const size_t out_elems = (size_t)BB * SS * 2 * HH;
  float* dout_h = out + out_elems;
  float* dout_c = dout_h + (size_t)4 * BB * HH;

  dim3 gg(GG / 128, (SS * BB) / 64);

  // ---- layer 0: input projections + recurrence ----
  wmma_gemm_bias_kernel<<<gg, TPB, 0, stream>>>(xbf, wih_p[0], biasc[0], xpf, SS * BB, GG, II);
  wmma_gemm_bias_kernel<<<gg, TPB, 0, stream>>>(xbf, wih_p[1], biasc[1], xpb, SS * BB, GG, II);

  zero_kernel<<<64, TPB, 0, stream>>>((u32*)hbf, (2 * BB * HH * 2) / 4);
  zero_kernel<<<64, TPB, 0, stream>>>((u32*)cst, 2 * BB * HH);
  zero_kernel<<<1, 64, 0, stream>>>(cnt, 64);
  lstm_layer_kernel<<<NWG, TPB, 0, stream>>>(xpf, xpb, whh_p[0], whh_p[1], gates, hbf, cst,
                                             seq1, nullptr, dout_h, dout_c, cnt);

  // ---- layer 1: input projections (from seq1) + recurrence ----
  wmma_gemm_bias_kernel<<<gg, TPB, 0, stream>>>(seq1, wih_p[2], biasc[2], xpf, SS * BB, GG, 2 * HH);
  wmma_gemm_bias_kernel<<<gg, TPB, 0, stream>>>(seq1, wih_p[3], biasc[3], xpb, SS * BB, GG, 2 * HH);

  zero_kernel<<<64, TPB, 0, stream>>>((u32*)hbf, (2 * BB * HH * 2) / 4);
  zero_kernel<<<64, TPB, 0, stream>>>((u32*)cst, 2 * BB * HH);
  zero_kernel<<<1, 64, 0, stream>>>(cnt, 64);
  lstm_layer_kernel<<<NWG, TPB, 0, stream>>>(xpf, xpb, whh_p[2], whh_p[3], gates, hbf, cst,
                                             nullptr, out, dout_h + 2 * BB * HH,
                                             dout_c + 2 * BB * HH, cnt);
}
